// WindowAttention_all_15178414424628
// MI455X (gfx1250) — matrix operand order repaired and checked by reading, not compiled, not run on hardware
//
#include <hip/hip_runtime.h>

// MI455X / gfx1250 WindowAttention (dual stream) — bf16 WMMA pipeline.
// Workspace layout (needs ~822 MB):
//   qkv_x  : M*768 bf16      qkv_rgb : M*768 bf16
//   ax     : M*256 bf16      ar      : M*256 bf16     (M = 4096*49 = 200704)

typedef __attribute__((ext_vector_type(16))) __bf16 v16bf;
typedef __attribute__((ext_vector_type(8)))  float  v8f;

union BF16Frag { v16bf v; unsigned short u[16]; };

__device__ __forceinline__ unsigned short f2bf(float f) {
  unsigned int x = __float_as_uint(f);
  x += 0x7FFFu + ((x >> 16) & 1u);   // round-to-nearest-even
  return (unsigned short)(x >> 16);
}

// ---------------------------------------------------------------------------
// Generic tiled GEMM:  C[M,N] = A[M,K] @ W[K,N] + bias[N]
// A is f32 (inputs) or bf16 (workspace); C is bf16 (workspace) or f32 (d_out).
// Block: 256 threads (8 waves). Tile 64x64, K-step 32. M,N multiples of 64.
// ---------------------------------------------------------------------------
template<bool A_IS_BF16, bool OUT_IS_F32>
__global__ __launch_bounds__(256) void gemm_wmma(
    const void* __restrict__ Aptr, const float* __restrict__ W,
    const float* __restrict__ bias, void* __restrict__ Cptr,
    int K, int N)
{
  __shared__ unsigned short As[64][32];   // As[m][k]
  __shared__ unsigned short Bs[64][32];   // Bs[n][k]  (W transposed tile)

  const int t    = threadIdx.x;
  const int lane = t & 31;
  const int wv   = t >> 5;
  const int m0   = blockIdx.y * 64;
  const int n0   = blockIdx.x * 64;
  const int tm   = wv >> 1;          // wave's M-tile (0..3)
  const int tn0  = (wv & 1) * 2;     // wave's first N-tile (0 or 2)
  const int g    = lane >> 4;
  const int lm   = lane & 15;

  const float*          Af = (const float*)Aptr;
  const unsigned short* Ab = (const unsigned short*)Aptr;

  v8f acc0 = {}; v8f acc1 = {};

  for (int k0 = 0; k0 < K; k0 += 32) {
    #pragma unroll
    for (int i = t; i < 64 * 32; i += 256) {
      int r = i >> 5, c = i & 31;
      size_t gi = (size_t)(m0 + r) * K + (k0 + c);
      As[r][c] = A_IS_BF16 ? Ab[gi] : f2bf(Af[gi]);
    }
    #pragma unroll
    for (int i = t; i < 64 * 32; i += 256) {
      int n = i >> 5, k = i & 31;
      Bs[n][k] = f2bf(W[(size_t)(k0 + k) * N + (n0 + n)]);
    }
    __syncthreads();

    // Prefetch next K-step tiles (global_prefetch_b8) to overlap with WMMA.
    if (k0 + 32 < K) {
      int pr = t >> 2, pc = (t & 3) * 8;            // A: 64 rows x 4 chunks
      if (A_IS_BF16) __builtin_prefetch(Ab + (size_t)(m0 + pr) * K + (k0 + 32 + pc), 0, 1);
      else           __builtin_prefetch(Af + (size_t)(m0 + pr) * K + (k0 + 32 + pc), 0, 1);
      int pk = t >> 3, pn = (t & 7) * 8;            // W: 32 k-rows x 8 chunks
      __builtin_prefetch(W + (size_t)(k0 + 32 + pk) * N + (n0 + pn), 0, 1);
    }

    BF16Frag a;
    #pragma unroll
    for (int j = 0; j < 8; ++j) {            // A layout: lanes hold row lm
      int kl = (j < 4) ? (8*g + 2*j) : (16 + 8*g + 2*(j - 4));
      a.u[2*j]   = As[tm*16 + lm][kl];
      a.u[2*j+1] = As[tm*16 + lm][kl + 1];
    }
    #pragma unroll
    for (int tt = 0; tt < 2; ++tt) {
      BF16Frag b;
      int cn = (tn0 + tt) * 16 + lm;         // B layout: lane = out column
      #pragma unroll
      for (int j = 0; j < 8; ++j) {
        int kl = (j < 4) ? (8*g + 2*j) : (16 + 8*g + 2*(j - 4));
        b.u[2*j]   = Bs[cn][kl];
        b.u[2*j+1] = Bs[cn][kl + 1];
      }
      if (tt == 0)
        acc0 = __builtin_amdgcn_wmma_f32_16x16x32_bf16(false, a.v, false, b.v,
                                                       (short)0, acc0, false, false);
      else
        acc1 = __builtin_amdgcn_wmma_f32_16x16x32_bf16(false, a.v, false, b.v,
                                                       (short)0, acc1, false, false);
    }
    __syncthreads();
  }

  #pragma unroll
  for (int tt = 0; tt < 2; ++tt) {
    int col = n0 + (tn0 + tt) * 16 + lm;
    float bc = bias[col];
    v8f acc = tt ? acc1 : acc0;
    #pragma unroll
    for (int j = 0; j < 8; ++j) {
      int row = m0 + tm*16 + j + 8*g;        // C layout: VGPR j -> row j+8g
      float val = acc[j] + bc;
      if (OUT_IS_F32) ((float*)Cptr)[(size_t)row * N + col] = val;
      else            ((unsigned short*)Cptr)[(size_t)row * N + col] = f2bf(val);
    }
  }
}

// ---------------------------------------------------------------------------
// Fused window attention: one wave per (window b, head h, stream s).
// stream 0: softmax(q_rgb.k^T*s + bias + mask) @ v        -> ax
// stream 1: softmax(q.k_rgb^T*s + bias + mask) @ v_rgb    -> ar
// 49 tokens padded to 64 (4 WMMA tiles); HD=32 = one bf16 K-step.
// ---------------------------------------------------------------------------
__global__ __launch_bounds__(32) void win_attn(
    const unsigned short* __restrict__ qkvx,
    const unsigned short* __restrict__ qkvr,
    const float* __restrict__ mask,
    const float* __restrict__ bias_table,
    const int* __restrict__ rel_idx,
    unsigned short* __restrict__ ax,
    unsigned short* __restrict__ ar)
{
  __shared__ unsigned short Qs[64][32];     // [token][d]
  __shared__ unsigned short Ks[64][32];     // [token][d]  (B operand of Q.K^T)
  __shared__ unsigned short VsT[32][64];    // [d][token]  (B operand of P.V)
  __shared__ float          BMt[64][64];    // [key col n][query row m]: bias+mask,
                                            //  -1e30 for n>=49, 0 for m>=49
  __shared__ unsigned short Ps[64][64];     // softmax probs, bf16

  const float SCALE = 0.1767766952966369f;  // 32^-0.5
  const int bid  = blockIdx.x;
  const int s    = bid >> 15;               // stream
  const int b    = (bid >> 3) & 4095;       // window
  const int h    = bid & 7;                 // head
  const int lane = threadIdx.x;
  const int wid  = b & 63;                  // mask window index (b % NW)
  const int g    = lane >> 4;
  const int lm   = lane & 15;

  const unsigned short* qb = (s == 0 ? qkvr : qkvx) + (size_t)b * 49 * 768 + h * 32;
  const unsigned short* kb = (s == 0 ? qkvx : qkvr) + (size_t)b * 49 * 768 + 256 + h * 32;
  const unsigned short* vb = kb + 256;
  unsigned short*       ob = (s == 0 ? ax : ar) + (size_t)b * 49 * 256 + h * 32;

  for (int i = lane; i < 64 * 32; i += 32) {
    int r = i >> 5, c = i & 31;
    unsigned short q = 0, k = 0, v = 0;
    if (r < 49) { q = qb[r*768 + c]; k = kb[r*768 + c]; v = vb[r*768 + c]; }
    Qs[r][c] = q; Ks[r][c] = k; VsT[c][r] = v;
  }
  // Bias + mask with padding baked in (transposed: consecutive m contiguous).
  for (int i = lane; i < 64 * 64; i += 32) {
    int n = i >> 6, m = i & 63;
    float v;
    if (n >= 49)      v = -1.0e30f;   // padded key column -> exp() == 0
    else if (m >= 49) v = 0.f;        // padded query row (result discarded)
    else v = bias_table[rel_idx[m*49 + n] * 8 + h] + mask[(size_t)wid * 2401 + m*49 + n];
    BMt[n][m] = v;
  }
  __syncthreads();

  // ---- S = Q K^T  (16 WMMAs) ----
  v8f S[4][4];
  #pragma unroll
  for (int tm = 0; tm < 4; ++tm) {
    BF16Frag a;
    #pragma unroll
    for (int j = 0; j < 8; ++j) {
      int kl = (j < 4) ? (8*g + 2*j) : (16 + 8*g + 2*(j - 4));
      a.u[2*j]   = Qs[tm*16 + lm][kl];
      a.u[2*j+1] = Qs[tm*16 + lm][kl + 1];
    }
    #pragma unroll
    for (int tn = 0; tn < 4; ++tn) {
      BF16Frag bf;
      int cn = tn*16 + lm;
      #pragma unroll
      for (int j = 0; j < 8; ++j) {
        int kl = (j < 4) ? (8*g + 2*j) : (16 + 8*g + 2*(j - 4));  // B[d][col] = K[col][d]
        bf.u[2*j]   = Ks[cn][kl];
        bf.u[2*j+1] = Ks[cn][kl + 1];
      }
      v8f z = {};
      S[tm][tn] = __builtin_amdgcn_wmma_f32_16x16x32_bf16(false, a.v, false, bf.v,
                                                          (short)0, z, false, false);
    }
  }

  // ---- S = S*scale + (bias+mask+padding), branch-free ----
  // Per (tm,tn) each lane reads 8 consecutive floats -> 2x ds_load_b128.
  #pragma unroll
  for (int tm = 0; tm < 4; ++tm)
    #pragma unroll
    for (int tn = 0; tn < 4; ++tn) {
      const float* bp = &BMt[tn*16 + lm][tm*16 + 8*g];
      #pragma unroll
      for (int j = 0; j < 8; ++j)
        S[tm][tn][j] = fmaf(S[tm][tn][j], SCALE, bp[j]);
    }

  // ---- row softmax: a row spans 16 lanes (one half-wave) x 4 tn tiles ----
  #pragma unroll
  for (int tm = 0; tm < 4; ++tm)
    #pragma unroll
    for (int j = 0; j < 8; ++j) {
      float rmax = -1.0e30f;
      #pragma unroll
      for (int tn = 0; tn < 4; ++tn) rmax = fmaxf(rmax, S[tm][tn][j]);
      #pragma unroll
      for (int o = 8; o >= 1; o >>= 1) rmax = fmaxf(rmax, __shfl_xor(rmax, o, 32));
      float rsum = 0.f;
      #pragma unroll
      for (int tn = 0; tn < 4; ++tn) {
        float e = __expf(S[tm][tn][j] - rmax);
        S[tm][tn][j] = e;
        rsum += e;
      }
      #pragma unroll
      for (int o = 8; o >= 1; o >>= 1) rsum += __shfl_xor(rsum, o, 32);
      float rinv = 1.0f / rsum;
      #pragma unroll
      for (int tn = 0; tn < 4; ++tn) S[tm][tn][j] *= rinv;
    }

  // ---- P -> LDS (bf16, A-operand for P.V) ----
  #pragma unroll
  for (int tm = 0; tm < 4; ++tm)
    #pragma unroll
    for (int tn = 0; tn < 4; ++tn)
      #pragma unroll
      for (int j = 0; j < 8; ++j)
        Ps[tm*16 + j + 8*g][tn*16 + lm] = f2bf(S[tm][tn][j]);
  __syncthreads();

  // ---- O = P V  (16 WMMAs: 4 tm x 2 td x 2 K-steps) ----
  #pragma unroll
  for (int tm = 0; tm < 4; ++tm) {
    #pragma unroll
    for (int td = 0; td < 2; ++td) {
      v8f o = {};
      #pragma unroll
      for (int ks = 0; ks < 2; ++ks) {
        BF16Frag a, bf;
        #pragma unroll
        for (int j = 0; j < 8; ++j) {
          int kl = ks*32 + ((j < 4) ? (8*g + 2*j) : (16 + 8*g + 2*(j - 4)));
          a.u[2*j]   = Ps[tm*16 + lm][kl];
          a.u[2*j+1] = Ps[tm*16 + lm][kl + 1];
        }
        #pragma unroll
        for (int j = 0; j < 8; ++j) {
          int kl = ks*32 + ((j < 4) ? (8*g + 2*j) : (16 + 8*g + 2*(j - 4)));  // B[n][d] = V[n][d], via VsT[d][n]
          bf.u[2*j]   = VsT[td*16 + lm][kl];
          bf.u[2*j+1] = VsT[td*16 + lm][kl + 1];
        }
        o = __builtin_amdgcn_wmma_f32_16x16x32_bf16(false, a.v, false, bf.v,
                                                    (short)0, o, false, false);
      }
      #pragma unroll
      for (int j = 0; j < 8; ++j) {
        int m = tm*16 + j + 8*g;
        if (m < 49) ob[(size_t)m * 256 + td*16 + lm] = f2bf(o[j]);
      }
    }
  }
}

// ---------------------------------------------------------------------------
extern "C" void kernel_launch(void* const* d_in, const int* in_sizes, int n_in,
                              void* d_out, int out_size, void* d_ws, size_t ws_size,
                              hipStream_t stream) {
  const float* x          = (const float*)d_in[0];
  const float* rgb        = (const float*)d_in[1];
  const float* mask       = (const float*)d_in[2];
  const float* w_qkv      = (const float*)d_in[3];
  const float* b_qkv      = (const float*)d_in[4];
  const float* w_qkv_rgb  = (const float*)d_in[5];
  const float* b_qkv_rgb  = (const float*)d_in[6];
  const float* bias_table = (const float*)d_in[7];
  const float* w_proj     = (const float*)d_in[8];
  const float* b_proj     = (const float*)d_in[9];
  const float* w_proj_rgb = (const float*)d_in[10];
  const float* b_proj_rgb = (const float*)d_in[11];
  const int*   rel_idx    = (const int*)d_in[12];

  const int M = 4096 * 49;                  // 200704 = 3136 * 64
  unsigned short* qkvx = (unsigned short*)d_ws;
  unsigned short* qkvr = qkvx + (size_t)M * 768;
  unsigned short* axw  = qkvr + (size_t)M * 768;
  unsigned short* arw  = axw  + (size_t)M * 256;

  float* xo = (float*)d_out;
  float* ro = xo + (size_t)M * 256;

  dim3 blk(256);
  // QKV projections (f32 in -> bf16 out)
  gemm_wmma<false, false><<<dim3(768 / 64, M / 64), blk, 0, stream>>>(
      x,   w_qkv,     b_qkv,     qkvx, 256, 768);
  gemm_wmma<false, false><<<dim3(768 / 64, M / 64), blk, 0, stream>>>(
      rgb, w_qkv_rgb, b_qkv_rgb, qkvr, 256, 768);

  // Fused attention: 4096 windows x 8 heads x 2 streams, one wave each
  win_attn<<<4096 * 8 * 2, 32, 0, stream>>>(qkvx, qkvr, mask, bias_table,
                                            rel_idx, axw, arw);

  // Output projections (bf16 in -> f32 out, straight into d_out)
  gemm_wmma<true, true><<<dim3(256 / 64, M / 64), blk, 0, stream>>>(
      axw, w_proj,     b_proj,     xo, 256, 256);
  gemm_wmma<true, true><<<dim3(256 / 64, M / 64), blk, 0, stream>>>(
      arw, w_proj_rgb, b_proj_rgb, ro, 256, 256);
}
